// TemporalDiffusionTransformerDecoderLayer_27169963114865
// MI455X (gfx1250) — matrix operand order repaired and checked by reading, not compiled, not run on hardware
//
#include <hip/hip_runtime.h>
#include <stddef.h>

// ---------------------------------------------------------------------------
// CDNA5 (gfx1250) wave32 WMMA implementation of the temporal transformer layer
// B=16, T=1024, D=512, H=8 (dh=64), FFN=2048, SE_HID=256
// ---------------------------------------------------------------------------

typedef __attribute__((ext_vector_type(16))) _Float16 v16h;
typedef __attribute__((ext_vector_type(8)))  float    v8f;
typedef __attribute__((ext_vector_type(4)))  int      v4i_t;

union FragU { uint4 q[2]; v16h v; };

static __device__ __forceinline__ v16h load_frag(const _Float16* p0, const _Float16* p1) {
    FragU u;
    u.q[0] = *reinterpret_cast<const uint4*>(p0);
    u.q[1] = *reinterpret_cast<const uint4*>(p1);
    return u.v;
}

// Async global -> LDS 16-byte copy (ASYNCcnt-tracked).
static __device__ __forceinline__ void async_b128(const void* g, void* l) {
#if __has_builtin(__builtin_amdgcn_global_load_async_to_lds_b128)
    __builtin_amdgcn_global_load_async_to_lds_b128(
        (__attribute__((address_space(1))) v4i_t*)g,
        (__attribute__((address_space(3))) v4i_t*)l, 0, 0);
#else
    const unsigned la = (unsigned)(size_t)l;   // generic LDS addr: low 32 bits
    asm volatile("global_load_async_to_lds_b128 %0, %1, off"
                 :: "v"(la), "v"(g) : "memory");
#endif
}

static __device__ __forceinline__ void wait_async0() {
#if __has_builtin(__builtin_amdgcn_s_wait_asynccnt)
    __builtin_amdgcn_s_wait_asynccnt(0);
#else
    asm volatile("s_wait_asynccnt 0" ::: "memory");
#endif
}

// ---------------------------------------------------------------------------
// SE gate: y = mean(x, -1)
// ---------------------------------------------------------------------------
__global__ __launch_bounds__(256)
void mean_kernel(const float* __restrict__ x, float* __restrict__ y) {
    const int lane = threadIdx.x & 31, wave = threadIdx.x >> 5;
    const int row  = blockIdx.x * 8 + wave;                 // 16384 rows
    const float* xr = x + (size_t)row * 512;
    float s = 0.f;
#pragma unroll
    for (int i = 0; i < 16; ++i) s += xr[lane + i * 32];
#pragma unroll
    for (int m = 16; m >= 1; m >>= 1) s += __shfl_xor(s, m, 32);
    if (lane == 0) y[row] = s * (1.0f / 512.0f);
}

// gate = sigmoid(relu(y @ se_w1) @ se_w2), one block per batch
__global__ __launch_bounds__(256)
void se_kernel(const float* __restrict__ y, const float* __restrict__ w1,
               const float* __restrict__ w2, float* __restrict__ gate) {
    __shared__ float ys[1024];
    __shared__ float hs[256];
    const int b = blockIdx.x, tid = threadIdx.x;
#pragma unroll
    for (int i = 0; i < 4; ++i) ys[tid + i * 256] = y[b * 1024 + tid + i * 256];
    __syncthreads();
    float acc = 0.f;
    for (int t = 0; t < 1024; ++t) acc += ys[t] * w1[t * 256 + tid];
    hs[tid] = fmaxf(acc, 0.f);
    __syncthreads();
#pragma unroll
    for (int i = 0; i < 4; ++i) {
        const int t = tid + i * 256;
        float g = 0.f;
        for (int j = 0; j < 256; ++j) g += hs[j] * w2[j * 1024 + t];
        gate[b * 1024 + t] = 1.0f / (1.0f + __expf(-g));
    }
}

// x2 = x * (1 + gate[row]) elementwise (float4 vectorized)
__global__ __launch_bounds__(256)
void segate_kernel(const float* __restrict__ x, const float* __restrict__ gate,
                   float* __restrict__ x2) {
    const size_t i = ((size_t)blockIdx.x * 256 + threadIdx.x) * 4;
    float4 v = *reinterpret_cast<const float4*>(x + i);
    const float g = 1.0f + gate[i >> 9];                    // row = i / 512
    float4 o = make_float4(v.x * g, v.y * g, v.z * g, v.w * g);
    *reinterpret_cast<float4*>(x2 + i) = o;
}

// ---------------------------------------------------------------------------
// LayerNorm over D=512, one block per row, writes f16
// ---------------------------------------------------------------------------
__global__ __launch_bounds__(256)
void ln_kernel(const float* __restrict__ x, const float* __restrict__ g,
               const float* __restrict__ be, _Float16* __restrict__ out) {
    __shared__ float red[256];
    __shared__ float mu_s, rs_s;
    const int row = blockIdx.x, tid = threadIdx.x;
    const float* xr = x + (size_t)row * 512;
    const float a = xr[tid], c = xr[tid + 256];
    red[tid] = a + c; __syncthreads();
    for (int s = 128; s > 0; s >>= 1) { if (tid < s) red[tid] += red[tid + s]; __syncthreads(); }
    if (tid == 0) mu_s = red[0] * (1.0f / 512.0f);
    __syncthreads();
    const float mu = mu_s;
    const float d0 = a - mu, d1 = c - mu;
    red[tid] = d0 * d0 + d1 * d1; __syncthreads();
    for (int s = 128; s > 0; s >>= 1) { if (tid < s) red[tid] += red[tid + s]; __syncthreads(); }
    if (tid == 0) rs_s = rsqrtf(red[0] * (1.0f / 512.0f) + 1e-5f);
    __syncthreads();
    const float rs = rs_s;
    out[(size_t)row * 512 + tid]       = (_Float16)(d0 * rs * g[tid] + be[tid]);
    out[(size_t)row * 512 + tid + 256] = (_Float16)(d1 * rs * g[tid + 256] + be[tid + 256]);
}

// ---------------------------------------------------------------------------
// Weight pre-pass: Wt[n][k] = (f16) W[k][n]   (W is KxN row-major)
// ---------------------------------------------------------------------------
__global__ __launch_bounds__(256)
void wtrans_kernel(const float* __restrict__ W, _Float16* __restrict__ Wt,
                   int K, int N) {
    const size_t i = (size_t)blockIdx.x * 256 + threadIdx.x;   // over N*K
    const int n = (int)(i / K), k = (int)(i % K);
    Wt[(size_t)n * K + k] = (_Float16)W[(size_t)k * N + n];
}

// ---------------------------------------------------------------------------
// Tiled WMMA GEMM: C(MxN) = A(MxK,f16,row-major) * Bt(NxK,f16,row-major = B^T)
// Block tile 128x64, 8 waves, each wave a 32x32 C tile (2x2 WMMA tiles), BK=32.
// Double-buffered LDS staged with GLOBAL_LOAD_ASYNC_TO_LDS_B128 (ASYNCcnt).
// MODE 0: scatter (B,H,T,64) f16 (Q)     MODE 1: same layout (K)
// MODE 2: scatter (B,H,64,T) f16 (V^T)   MODE 3: GELU(.+b1) -> f16 row-major
// MODE 4: out = resid + C + bias  (f32)
// ---------------------------------------------------------------------------
template<int MODE>
__global__ __launch_bounds__(256)
void gemm_f16(const _Float16* __restrict__ A, const _Float16* __restrict__ Bt,
              const float* __restrict__ bias, const float* __restrict__ resid,
              float* __restrict__ outF, _Float16* __restrict__ outH,
              int M, int N, int K) {
    __shared__ _Float16 Al[2][128][40];   // row-major [m][k], +8 halves pad
    __shared__ _Float16 Bl[2][64][40];    // transposed [n][k]
    const int tid  = threadIdx.x;
    const int lane = tid & 31;
    const int wave = tid >> 5;
    const int wm = wave & 3, wn = wave >> 2;
    const int m0 = blockIdx.y * 128, n0 = blockIdx.x * 64;
    const int lo    = lane & 15;
    const int hi8   = (lane >> 4) << 3;             // C/D row offset 0|8
    const int abase = (lane < 16) ? 0 : 8;          // A-frag K chunk base
    const int kb16  = (lane < 16) ? 0 : 16;         // unused (B frag also uses abase)

    v8f acc[2][2] = {};

    const int ar = tid >> 1, aseg = (tid & 1) * 16;     // A staging: 2 thr/row
    const int bn = tid >> 2, bseg = (tid & 3) * 8;      // B staging: 4 thr/row

    auto stage = [&](int kt, int buf) {
        const int k0 = kt * 32;
        const _Float16* as = A + (size_t)(m0 + ar) * K + k0 + aseg;
        async_b128(as,     &Al[buf][ar][aseg]);
        async_b128(as + 8, &Al[buf][ar][aseg + 8]);
        const _Float16* bs = Bt + (size_t)(n0 + bn) * K + k0 + bseg;
        async_b128(bs, &Bl[buf][bn][bseg]);
    };

    const int nk = K >> 5;
    stage(0, 0);                                     // prologue
    for (int kt = 0; kt < nk; ++kt) {
        const int cur = kt & 1;
        wait_async0();                               // this wave's copies landed
        __syncthreads();                             // everyone's copies landed

        v16h af[2], bf[2];
#pragma unroll
        for (int i = 0; i < 2; ++i) {
            const int m = wm * 32 + i * 16 + lo;
            af[i] = load_frag(&Al[cur][m][abase], &Al[cur][m][abase + 16]);
        }
#pragma unroll
        for (int j = 0; j < 2; ++j) {
            const int n = wn * 32 + j * 16 + lo;
            bf[j] = load_frag(&Bl[cur][n][abase], &Bl[cur][n][abase + 16]);
        }
        // kick off next tile's copies; they overlap with the WMMAs below
        if (kt + 1 < nk) stage(kt + 1, cur ^ 1);

#pragma unroll
        for (int i = 0; i < 2; ++i)
#pragma unroll
            for (int j = 0; j < 2; ++j)
                acc[i][j] = __builtin_amdgcn_wmma_f32_16x16x32_f16(
                    false, af[i], false, bf[j], (short)0, acc[i][j], false, false);
    }

#pragma unroll
    for (int i = 0; i < 2; ++i)
#pragma unroll
        for (int j = 0; j < 2; ++j)
#pragma unroll
            for (int r = 0; r < 8; ++r) {
                const int row = m0 + wm * 32 + i * 16 + r + hi8;
                const int col = n0 + wn * 32 + j * 16 + lo;
                const float v = acc[i][j][r];
                if (MODE <= 2) {
                    const int b = row >> 10, t = row & 1023;
                    const int h = col >> 6,  d = col & 63;
                    if (MODE == 2)
                        outH[(((size_t)(b * 8 + h)) * 64 + d) * 1024 + t] = (_Float16)v;
                    else
                        outH[(((size_t)(b * 8 + h)) * 1024 + t) * 64 + d] = (_Float16)v;
                } else if (MODE == 3) {
                    const float z = v + bias[col];
                    const float gl = 0.5f * z * (1.0f + erff(z * 0.70710678118f));
                    outH[(size_t)row * N + col] = (_Float16)gl;
                } else {
                    outF[(size_t)row * N + col] =
                        resid[(size_t)row * N + col] + v + bias[col];
                }
            }
}

// ---------------------------------------------------------------------------
// Flash attention: one block = 128 query rows of one (b,h); wave owns 16 rows.
// Q/K (B,H,T,64) f16, Vt (B,H,64,T) f16. Writes x3 = x2 + attn_out (f32).
// ---------------------------------------------------------------------------
__global__ __launch_bounds__(256)
void attn_kernel(const _Float16* __restrict__ Q, const _Float16* __restrict__ Kb,
                 const _Float16* __restrict__ Vt, const float* __restrict__ x2,
                 float* __restrict__ x3) {
    __shared__ _Float16 Pl[8][16][40];   // per-wave P tile [row][key], padded
    const int tid = threadIdx.x, lane = tid & 31, wave = tid >> 5;
    const int bh = blockIdx.y;                      // b*8 + h
    const int b = bh >> 3, h = bh & 7;
    const int q0 = blockIdx.x * 128 + wave * 16;
    const int lo    = lane & 15;
    const int hi8   = (lane >> 4) << 3;
    const int abase = (lane < 16) ? 0 : 8;
    const int kb16  = (lane < 16) ? 0 : 16;

    const _Float16* Qh = Q  + (size_t)bh * 1024 * 64;
    const _Float16* Kh = Kb + (size_t)bh * 1024 * 64;
    const _Float16* Vh = Vt + (size_t)bh * 64 * 1024;

    // Q fragments for this wave's 16 rows (head dim 64 -> 2 k-steps)
    v16h qf[2];
#pragma unroll
    for (int ks = 0; ks < 2; ++ks) {
        const _Float16* p = Qh + (size_t)(q0 + lo) * 64 + ks * 32;
        qf[ks] = load_frag(p + abase, p + abase + 16);
    }

    v8f O[4] = {};
    float mrow[8], lrow[8];
#pragma unroll
    for (int r = 0; r < 8; ++r) { mrow[r] = -3.0e38f; lrow[r] = 0.0f; }

    for (int kv = 0; kv < 1024; kv += 32) {
        // scores S (16 x 32) = Q * K^T : K rows are exactly B-fragment layout
        v8f S[2] = {};
#pragma unroll
        for (int t = 0; t < 2; ++t)
#pragma unroll
            for (int ks = 0; ks < 2; ++ks) {
                const _Float16* p = Kh + (size_t)(kv + t * 16 + lo) * 64 + ks * 32 + abase;
                v16h bfr = load_frag(p, p + 16);
                S[t] = __builtin_amdgcn_wmma_f32_16x16x32_f16(
                    false, qf[ks], false, bfr, (short)0, S[t], false, false);
            }
        // online softmax over this 32-key block
        float alpha[8];
#pragma unroll
        for (int r = 0; r < 8; ++r) {
            const float s0 = S[0][r] * 0.125f, s1 = S[1][r] * 0.125f;
            float mx = fmaxf(s0, s1);
#pragma unroll
            for (int mm = 8; mm >= 1; mm >>= 1) mx = fmaxf(mx, __shfl_xor(mx, mm, 32));
            const float mnew = fmaxf(mrow[r], mx);
            alpha[r] = __expf(mrow[r] - mnew);
            mrow[r] = mnew;
            const float p0 = __expf(s0 - mnew), p1 = __expf(s1 - mnew);
            Pl[wave][r + hi8][lo]      = (_Float16)p0;
            Pl[wave][r + hi8][16 + lo] = (_Float16)p1;
            float ps = p0 + p1;
#pragma unroll
            for (int mm = 8; mm >= 1; mm >>= 1) ps += __shfl_xor(ps, mm, 32);
            lrow[r] = lrow[r] * alpha[r] + ps;
        }
#pragma unroll
        for (int t = 0; t < 4; ++t)
#pragma unroll
            for (int r = 0; r < 8; ++r) O[t][r] *= alpha[r];

        // P tile written by all 32 lanes -> wait for LDS before fragment load
        asm volatile("s_wait_dscnt 0" ::: "memory");
        v16h pf = load_frag(&Pl[wave][lo][abase], &Pl[wave][lo][abase + 16]);
        // O += P * V : Vt rows are exactly B-fragment layout
#pragma unroll
        for (int t = 0; t < 4; ++t) {
            const _Float16* p = Vh + (size_t)(t * 16 + lo) * 1024 + kv + abase;
            v16h vfr = load_frag(p, p + 16);
            O[t] = __builtin_amdgcn_wmma_f32_16x16x32_f16(
                false, pf, false, vfr, (short)0, O[t], false, false);
        }
    }

    // x3 = x2 + O / l  (fused residual)
#pragma unroll
    for (int t = 0; t < 4; ++t)
#pragma unroll
        for (int r = 0; r < 8; ++r) {
            const int row = q0 + r + hi8;
            const int d = t * 16 + lo;
            const size_t idx = ((size_t)b * 1024 + row) * 512 + h * 64 + d;
            x3[idx] = x2[idx] + O[t][r] / lrow[r];
        }
}

// ---------------------------------------------------------------------------
extern "C" void kernel_launch(void* const* d_in, const int* in_sizes, int n_in,
                              void* d_out, int out_size, void* d_ws, size_t ws_size,
                              hipStream_t stream) {
    (void)in_sizes; (void)n_in; (void)out_size; (void)ws_size;
    const float* x    = (const float*)d_in[0];
    const float* wq   = (const float*)d_in[1];
    const float* wk   = (const float*)d_in[2];
    const float* wv   = (const float*)d_in[3];
    const float* ln1g = (const float*)d_in[4];
    const float* ln1b = (const float*)d_in[5];
    const float* sew1 = (const float*)d_in[6];
    const float* sew2 = (const float*)d_in[7];
    const float* ffng = (const float*)d_in[8];
    const float* ffnb = (const float*)d_in[9];
    const float* w1   = (const float*)d_in[10];
    const float* b1   = (const float*)d_in[11];
    const float* w2   = (const float*)d_in[12];
    const float* b2   = (const float*)d_in[13];
    float* out = (float*)d_out;

    char* ws = (char*)d_ws;
    size_t off = 0;
    auto alloc = [&](size_t bytes) -> void* {
        off = (off + 255) & ~(size_t)255;
        void* p = ws + off;
        off += bytes;
        return p;
    };
    float*    y    = (float*)alloc((size_t)16384 * 4);
    float*    gate = (float*)alloc((size_t)16384 * 4);
    float*    x2   = (float*)alloc((size_t)16384 * 512 * 4);
    float*    x3   = (float*)alloc((size_t)16384 * 512 * 4);
    _Float16* nb   = (_Float16*)alloc((size_t)16384 * 512 * 2);   // n, reused as n2
    _Float16* qb   = (_Float16*)alloc((size_t)16384 * 512 * 2);
    _Float16* kb   = (_Float16*)alloc((size_t)16384 * 512 * 2);
    _Float16* vtb  = (_Float16*)alloc((size_t)16384 * 512 * 2);
    _Float16* hb   = (_Float16*)alloc((size_t)16384 * 2048 * 2);
    _Float16* wqT  = (_Float16*)alloc((size_t)512 * 512 * 2);
    _Float16* wkT  = (_Float16*)alloc((size_t)512 * 512 * 2);
    _Float16* wvT  = (_Float16*)alloc((size_t)512 * 512 * 2);
    _Float16* w1T  = (_Float16*)alloc((size_t)512 * 2048 * 2);
    _Float16* w2T  = (_Float16*)alloc((size_t)2048 * 512 * 2);

    // weight pre-pass (f16, transposed -> pure-copy async staging in GEMMs)
    wtrans_kernel<<<1024, 256, 0, stream>>>(wq, wqT, 512, 512);
    wtrans_kernel<<<1024, 256, 0, stream>>>(wk, wkT, 512, 512);
    wtrans_kernel<<<1024, 256, 0, stream>>>(wv, wvT, 512, 512);
    wtrans_kernel<<<4096, 256, 0, stream>>>(w1, w1T, 512, 2048);
    wtrans_kernel<<<4096, 256, 0, stream>>>(w2, w2T, 2048, 512);

    // SE gate
    mean_kernel<<<2048, 256, 0, stream>>>(x, y);
    se_kernel<<<16, 256, 0, stream>>>(y, sew1, sew2, gate);
    segate_kernel<<<8192, 256, 0, stream>>>(x, gate, x2);

    // LN1 + QKV projections (f16 WMMA, async-staged)
    ln_kernel<<<16384, 256, 0, stream>>>(x2, ln1g, ln1b, nb);
    dim3 g512(8, 128);
    gemm_f16<0><<<g512, 256, 0, stream>>>(nb, wqT, nullptr, nullptr, nullptr, qb, 16384, 512, 512);
    gemm_f16<1><<<g512, 256, 0, stream>>>(nb, wkT, nullptr, nullptr, nullptr, kb, 16384, 512, 512);
    gemm_f16<2><<<g512, 256, 0, stream>>>(nb, wvT, nullptr, nullptr, nullptr, vtb, 16384, 512, 512);

    // attention + residual
    attn_kernel<<<dim3(8, 128), 256, 0, stream>>>(qb, kb, vtb, x2, x3);

    // LN2 + FFN
    ln_kernel<<<16384, 256, 0, stream>>>(x3, ffng, ffnb, nb);
    gemm_f16<3><<<dim3(32, 128), 256, 0, stream>>>(nb, w1T, b1, nullptr, nullptr, hb, 16384, 2048, 512);
    gemm_f16<4><<<dim3(8, 128), 256, 0, stream>>>(hb, w2T, b2, x3, out, nullptr, 16384, 512, 2048);
}
